// CoreAttention_59863254172126
// MI455X (gfx1250) — compile-verified, matrix operand order checked by reading, not run on hardware
//
#include <hip/hip_runtime.h>
#include <hip/hip_bf16.h>

typedef __attribute__((ext_vector_type(16))) _Float16 v16h;
typedef __attribute__((ext_vector_type(8)))  _Float16 h8;
typedef __attribute__((ext_vector_type(8)))  float    v8f;

#define S_TOTAL 4096
#define NH      16
#define DH      128
#define KV_TILE 64    // 4 column tiles of 16
#define Q_TILE  128   // 8 waves * 16 rows
#define KPAD    136   // 128 + 8 halfs  (272B rows, 16B aligned, 4-bank rotation)
#define VPAD    72    // 64 + 8 halfs   (144B rows, 16B aligned)
#define PPAD    72

// 16-lane rotation-reduce via DPP16 row_ror (stays inside each 16-lane row,
// matching the C-layout half-wave groups). Pure VALU — no LDS traffic.
// DPP ctrl must be a literal constant, so the 4 stages are written out.
__device__ __forceinline__ float dpp_ror1(float x) {
  return __int_as_float(__builtin_amdgcn_update_dpp(0, __float_as_int(x), 0x121, 0xf, 0xf, true));
}
__device__ __forceinline__ float dpp_ror2(float x) {
  return __int_as_float(__builtin_amdgcn_update_dpp(0, __float_as_int(x), 0x122, 0xf, 0xf, true));
}
__device__ __forceinline__ float dpp_ror4(float x) {
  return __int_as_float(__builtin_amdgcn_update_dpp(0, __float_as_int(x), 0x124, 0xf, 0xf, true));
}
__device__ __forceinline__ float dpp_ror8(float x) {
  return __int_as_float(__builtin_amdgcn_update_dpp(0, __float_as_int(x), 0x128, 0xf, 0xf, true));
}
__device__ __forceinline__ float rowred_max(float x) {
  x = fmaxf(x, dpp_ror1(x));
  x = fmaxf(x, dpp_ror2(x));
  x = fmaxf(x, dpp_ror4(x));
  x = fmaxf(x, dpp_ror8(x));
  return x;
}
__device__ __forceinline__ float rowred_sum(float x) {
  x += dpp_ror1(x);
  x += dpp_ror2(x);
  x += dpp_ror4(x);
  x += dpp_ror8(x);
  return x;
}

__device__ __forceinline__ h8 pack8(float4 a, float4 b) {
  h8 r;
  r[0] = (_Float16)a.x; r[1] = (_Float16)a.y;
  r[2] = (_Float16)a.z; r[3] = (_Float16)a.w;
  r[4] = (_Float16)b.x; r[5] = (_Float16)b.y;
  r[6] = (_Float16)b.z; r[7] = (_Float16)b.w;
  return r;
}

__global__ __launch_bounds__(256)
void ring_attn_fwd_f16wmma(const float* __restrict__ Q,
                           const float* __restrict__ K,
                           const float* __restrict__ V,
                           float* __restrict__ O) {
  __shared__ alignas(16) _Float16 Kl[KV_TILE][KPAD];   // K tile, row-major [kv][d]
  __shared__ alignas(16) _Float16 Vt[DH][VPAD];        // V tile, transposed [d][kv]
  __shared__ alignas(16) _Float16 Pl[8][16][PPAD];     // per-wave P bounce [m][kv]

  const int h    = blockIdx.y;
  const int q0   = blockIdx.x * Q_TILE;
  const int tid  = threadIdx.x;
  const int w    = tid >> 5;        // wave id (wave32)
  const int lane = tid & 31;
  const int n    = lane & 15;       // column / row-in-tile index
  const int hi   = lane >> 4;       // half-wave select

  // 1/sqrt(128) * log2(e): run softmax in base-2 (v_exp_f32)
  const float scale = 0.08838834764831845f * 1.4426950408889634f;

  // ---- preload this wave's 16 Q rows into WMMA A-layout (f16, pre-scaled) ----
  // A[m][kk]: lane m = lane&15; half j=8g+o  <->  kk = 16g + 8*hi + o
  v16h aq[4];
  {
    const int qrow = q0 + w * 16 + n;
    const float* qp = Q + ((size_t)qrow * NH + h) * DH;
#pragma unroll
    for (int c = 0; c < 4; ++c) {
#pragma unroll
      for (int g = 0; g < 2; ++g) {
        const float* src = qp + 32 * c + 16 * g + 8 * hi;
#pragma unroll
        for (int o = 0; o < 8; ++o)
          aq[c][8 * g + o] = (_Float16)(src[o] * scale);
      }
    }
  }

  v8f acc[8] = {};                  // 16x128 f32 accumulator, C-layout per d-chunk
  float mrow[8], lrow[8];
#pragma unroll
  for (int r = 0; r < 8; ++r) { mrow[r] = -__builtin_inff(); lrow[r] = 0.0f; }

  const int kvrow = tid >> 2;       // 0..63 : staging row
  const int cb    = (tid & 3) * 32; // 0..96 : staging col block (32 floats)

  for (int kv0 = 0; kv0 < S_TOTAL; kv0 += KV_TILE) {
    // ---- stage K (row-major f16, packed b128 stores) and V (transposed) ----
    {
      const float* kp = K + ((size_t)(kv0 + kvrow) * NH + h) * DH + cb;
      const float* vp = V + ((size_t)(kv0 + kvrow) * NH + h) * DH + cb;
      float4 k4[8], v4[8];
#pragma unroll
      for (int i = 0; i < 8; ++i) {
        k4[i] = ((const float4*)kp)[i];
        v4[i] = ((const float4*)vp)[i];
      }
#pragma unroll
      for (int i = 0; i < 4; ++i)
        *(h8*)&Kl[kvrow][cb + 8 * i] = pack8(k4[2*i], k4[2*i + 1]);
#pragma unroll
      for (int i = 0; i < 8; ++i) {
        Vt[cb + 4*i + 0][kvrow] = (_Float16)v4[i].x;
        Vt[cb + 4*i + 1][kvrow] = (_Float16)v4[i].y;
        Vt[cb + 4*i + 2][kvrow] = (_Float16)v4[i].z;
        Vt[cb + 4*i + 3][kvrow] = (_Float16)v4[i].w;
      }
      if (kv0 + KV_TILE < S_TOTAL) {          // global_prefetch_b8 next tile
        __builtin_prefetch(kp + (size_t)KV_TILE * NH * DH, 0, 0);
        __builtin_prefetch(vp + (size_t)KV_TILE * NH * DH, 0, 0);
      }
    }
    __syncthreads();

    // ---- scores S[16q x 64kv] = Q · K^T via 16 WMMAs ----
    v8f cs[4] = {};
#pragma unroll
    for (int c = 0; c < 4; ++c) {             // K-dim chunk of 32
#pragma unroll
      for (int cc = 0; cc < 4; ++cc) {        // kv column tile of 16
        h8 blo = *(const h8*)&Kl[n + 16*cc][32*c +      8*hi];
        h8 bhi = *(const h8*)&Kl[n + 16*cc][32*c + 16 + 8*hi];
        v16h b = __builtin_shufflevector(blo, bhi, 0,1,2,3,4,5,6,7,8,9,10,11,12,13,14,15);
        cs[cc] = __builtin_amdgcn_wmma_f32_16x16x32_f16(false, aq[c], false, b,
                                                        (short)0, cs[cc], false, false);
      }
    }

    // ---- online softmax (base-2), DPP reductions, P written to LDS bounce ----
    _Float16 (*Pw)[PPAD] = Pl[w];
    float sc[8];
#pragma unroll
    for (int r = 0; r < 8; ++r) {
      float s0 = cs[0][r], s1 = cs[1][r], s2 = cs[2][r], s3 = cs[3][r];
      float mx = fmaxf(fmaxf(s0, s1), fmaxf(s2, s3));
      mx = rowred_max(mx);
      float mnew = fmaxf(mrow[r], mx);
      sc[r] = __builtin_exp2f(mrow[r] - mnew);
      float p0 = __builtin_exp2f(s0 - mnew);
      float p1 = __builtin_exp2f(s1 - mnew);
      float p2 = __builtin_exp2f(s2 - mnew);
      float p3 = __builtin_exp2f(s3 - mnew);
      float sum = rowred_sum((p0 + p1) + (p2 + p3));
      lrow[r] = lrow[r] * sc[r] + sum;
      mrow[r] = mnew;
      Pw[8*hi + r][n     ] = (_Float16)p0;
      Pw[8*hi + r][n + 16] = (_Float16)p1;
      Pw[8*hi + r][n + 32] = (_Float16)p2;
      Pw[8*hi + r][n + 48] = (_Float16)p3;
    }
#pragma unroll
    for (int dc = 0; dc < 8; ++dc)
#pragma unroll
      for (int r = 0; r < 8; ++r)
        acc[dc][r] *= sc[r];

    asm volatile("s_wait_dscnt 0x0" ::: "memory");   // CDNA5 split DS counter fence

    // ---- O += P · V via 16 WMMAs against transposed-V LDS ----
#pragma unroll
    for (int t = 0; t < 2; ++t) {             // kv K-dim chunk of 32 for PV
      h8 plo = *(const h8*)&Pw[n][32*t +      8*hi];
      h8 phi = *(const h8*)&Pw[n][32*t + 16 + 8*hi];
      v16h ap = __builtin_shufflevector(plo, phi, 0,1,2,3,4,5,6,7,8,9,10,11,12,13,14,15);
#pragma unroll
      for (int dc = 0; dc < 8; ++dc) {
        h8 blo = *(const h8*)&Vt[dc*16 + n][32*t +      8*hi];
        h8 bhi = *(const h8*)&Vt[dc*16 + n][32*t + 16 + 8*hi];
        v16h bv = __builtin_shufflevector(blo, bhi, 0,1,2,3,4,5,6,7,8,9,10,11,12,13,14,15);
        acc[dc] = __builtin_amdgcn_wmma_f32_16x16x32_f16(false, ap, false, bv,
                                                         (short)0, acc[dc], false, false);
      }
    }
    __syncthreads();
  }

  // ---- finalize: divide by l, write out (coalesced across the 16-lane group) ----
#pragma unroll
  for (int r = 0; r < 8; ++r) {
    const int qrow = q0 + w * 16 + 8 * hi + r;
    const float recip = 1.0f / lrow[r];
    float* op = O + ((size_t)qrow * NH + h) * DH;
#pragma unroll
    for (int dc = 0; dc < 8; ++dc)
      op[dc * 16 + n] = acc[dc][r] * recip;
  }
}

extern "C" void kernel_launch(void* const* d_in, const int* in_sizes, int n_in,
                              void* d_out, int out_size, void* d_ws, size_t ws_size,
                              hipStream_t stream) {
  (void)in_sizes; (void)n_in; (void)d_ws; (void)ws_size; (void)out_size;
  const float* q = (const float*)d_in[0];
  const float* k = (const float*)d_in[1];
  const float* v = (const float*)d_in[2];
  // d_in[3] = ring_size: the sigmoid/log-sigmoid ring merge is algebraically
  // the online-softmax merge, so the continuous flash loop computes it exactly.
  float* out = (float*)d_out;
  dim3 grid(S_TOTAL / Q_TILE, NH);
  dim3 block(256);
  ring_attn_fwd_f16wmma<<<grid, block, 0, stream>>>(q, k, v, out);
}
